// QLSTMModel_86423331930414
// MI455X (gfx1250) — matrix operand order repaired and checked by reading, not compiled, not run on hardware
//
#include <hip/hip_runtime.h>

typedef __attribute__((ext_vector_type(16))) _Float16 v16h;
typedef __attribute__((ext_vector_type(8)))  _Float16 v8h;
typedef __attribute__((ext_vector_type(8)))  float    v8f;

#define B_SZ   512
#define T_SZ   1024
#define H_SZ   64
#define BTILE  16      // batch rows per workgroup
#define NTHREADS 128   // 4 wave32s

__device__ __forceinline__ float sigm_(float x) {
    return 1.0f / (1.0f + __expf(-x));
}
__device__ __forceinline__ float tanh_(float x) {
    // stable: ->  1 for large x, -1 for very negative x
    return 1.0f - 2.0f / (__expf(2.0f * x) + 1.0f);
}

__global__ __launch_bounds__(NTHREADS, 1)
void qlstm_wmma_kernel(const float* __restrict__ x,
                       const float* __restrict__ W_ih,
                       const float* __restrict__ W_hh,
                       const float* __restrict__ b_ih,
                       const float* __restrict__ b_hh,
                       const float* __restrict__ W_lin,
                       const float* __restrict__ b_lin,
                       float* __restrict__ out)
{
    // 32 KB x-tile (f16) + 4 KB double-buffered h tile -> fully LDS-resident recurrence
    __shared__ __align__(16) _Float16 xs[BTILE * T_SZ];
    __shared__ __align__(16) _Float16 hbuf[2][BTILE * H_SZ];

    const int tid  = threadIdx.x;
    const int wave = tid >> 5;        // 0..3 : which 16 hidden units this wave owns
    const int lane = tid & 31;
    const int lo16 = lane & 15;
    const int hi   = lane >> 4;       // 0 or 1 (lane half)
    const int bb   = blockIdx.x * BTILE;

    // ---- stage x[bb:bb+16, :] into LDS as f16 (streamed once; HBM side is trivial) ----
    for (int i = tid; i < BTILE * T_SZ / 4; i += NTHREADS) {
        const int row  = i >> 8;            // i / (T/4)
        const int col4 = i & 255;
        const float4 v = *(const float4*)(x + (size_t)(bb + row) * T_SZ + col4 * 4);
        _Float16* d = xs + row * T_SZ + col4 * 4;
        d[0] = (_Float16)v.x; d[1] = (_Float16)v.y;
        d[2] = (_Float16)v.z; d[3] = (_Float16)v.w;
    }
    // h0 = 0
    for (int i = tid; i < BTILE * H_SZ; i += NTHREADS) {
        hbuf[0][i] = (_Float16)0.0f;
        hbuf[1][i] = (_Float16)0.0f;
    }

    // ---- preload per-wave W_hh^T B-fragments (f32 -> f16), bias, W_ih column ----
    // Gate g of this wave occupies gate columns col = g*64 + wave*16 + lo16.
    // B layout (16x16x32 f16): lane holds column 'col', K = [32*kb + 8*hi, +8) then [32*kb + 16 + 8*hi, +8).
    v16h Bfrag[4][2];
    float biasv[4], wihv[4];
    #pragma unroll
    for (int g = 0; g < 4; ++g) {
        const int col = g * H_SZ + wave * 16 + lo16;      // 0..255
        biasv[g] = b_ih[col] + b_hh[col];
        wihv[g]  = W_ih[col];                              // IN == 1
        #pragma unroll
        for (int kb = 0; kb < 2; ++kb) {
            const int k0 = kb * 32 + hi * 8;
            const float* wp = W_hh + (size_t)col * H_SZ + k0;
            v16h b;
            #pragma unroll
            for (int e = 0; e < 16; ++e) b[e] = (_Float16)wp[e < 8 ? e : e + 8];
            Bfrag[g][kb] = b;
        }
    }

    __syncthreads();

    v8f cstate = {};   // cell state lives in registers, D-tile layout (r -> M=r+8*hi, lane -> N)

    for (int t = 0; t < T_SZ; ++t) {
        const int cur = t & 1;
        const _Float16* hc = hbuf[cur];

        // A fragments (h tile, 16x64 split into two 16x32 K-blocks)
        // A layout per lane: row M = lo16; halves = K[8*hi .. +8) then K[8*hi+16 .. +8)
        const v8h* ap = (const v8h*)(hc + lo16 * H_SZ);
        const v8h a0lo = ap[hi];
        const v8h a0hi = ap[hi + 2];
        const v8h a1lo = ap[hi + 4];
        const v8h a1hi = ap[hi + 6];
        const v16h A0 = __builtin_shufflevector(a0lo, a0hi, 0,1,2,3,4,5,6,7,8,9,10,11,12,13,14,15);
        const v16h A1 = __builtin_shufflevector(a1lo, a1hi, 0,1,2,3,4,5,6,7,8,9,10,11,12,13,14,15);

        // x_t for this lane's 8 batch rows (D-tile rows M = 8*hi + r)
        float xr[8];
        #pragma unroll
        for (int r = 0; r < 8; ++r)
            xr[r] = (float)xs[(hi * 8 + r) * T_SZ + t];

        // gates = h @ W_hh^T + (bias + x_t * W_ih) folded into the WMMA C input
        v8f gate[4];
        #pragma unroll
        for (int g = 0; g < 4; ++g) {
            v8f c0;
            #pragma unroll
            for (int r = 0; r < 8; ++r) c0[r] = __builtin_fmaf(xr[r], wihv[g], biasv[g]);
            c0 = __builtin_amdgcn_wmma_f32_16x16x32_f16(
                     false, A0, false, Bfrag[g][0], (short)0, c0, false, false);
            c0 = __builtin_amdgcn_wmma_f32_16x16x32_f16(
                     false, A1, false, Bfrag[g][1], (short)0, c0, false, false);
            gate[g] = c0;
        }

        // elementwise LSTM cell update; scatter h_new (f16) into the other h buffer
        _Float16* hn = hbuf[cur ^ 1];
        const int col = wave * 16 + lo16;
        #pragma unroll
        for (int r = 0; r < 8; ++r) {
            const float iv = sigm_(gate[0][r]);
            const float fv = sigm_(gate[1][r]);
            const float gv = tanh_(gate[2][r]);
            const float ov = sigm_(gate[3][r]);
            const float cn = fv * cstate[r] + iv * gv;
            cstate[r] = cn;
            const float hv = ov * tanh_(cn);
            const int m = hi * 8 + r;
            hn[m * H_SZ + col] = (_Float16)hv;
        }
        __syncthreads();   // one barrier per step (double-buffered h)
    }

    // ---- head: out[b] = h_final[b] . W_lin + b_lin  (16 outputs per workgroup) ----
    const _Float16* hf = hbuf[T_SZ & 1];
    if (tid < BTILE) {
        float acc = 0.0f;
        #pragma unroll 8
        for (int k = 0; k < H_SZ; ++k)
            acc = __builtin_fmaf((float)hf[tid * H_SZ + k], W_lin[k], acc);
        out[bb + tid] = acc + b_lin[0];
    }
}

extern "C" void kernel_launch(void* const* d_in, const int* in_sizes, int n_in,
                              void* d_out, int out_size, void* d_ws, size_t ws_size,
                              hipStream_t stream) {
    (void)in_sizes; (void)n_in; (void)d_ws; (void)ws_size; (void)out_size;
    const float* x     = (const float*)d_in[0];
    const float* W_ih  = (const float*)d_in[1];
    const float* W_hh  = (const float*)d_in[2];
    const float* b_ih  = (const float*)d_in[3];
    const float* b_hh  = (const float*)d_in[4];
    const float* W_lin = (const float*)d_in[5];
    const float* b_lin = (const float*)d_in[6];
    float* out = (float*)d_out;

    dim3 grid(B_SZ / BTILE);   // 32 workgroups, one 16-row batch tile each
    dim3 block(NTHREADS);      // 4 wave32s: wave w owns hidden units [16w, 16w+16)
    qlstm_wmma_kernel<<<grid, block, 0, stream>>>(x, W_ih, W_hh, b_ih, b_hh,
                                                  W_lin, b_lin, out);
}
